// VQVAE_76184129896749
// MI455X (gfx1250) — compile-verified, matrix operand order checked by reading, not run on hardware
//
#include <hip/hip_runtime.h>
#include <hip/hip_bf16.h>
#include <cstdint>
#include <cstddef>

typedef __attribute__((ext_vector_type(16))) _Float16 v16h;
typedef __attribute__((ext_vector_type(8)))  _Float16 v8h;
typedef __attribute__((ext_vector_type(8)))  float    v8f;
typedef __attribute__((ext_vector_type(4)))  unsigned int u32x4;
typedef __attribute__((ext_vector_type(8)))  int       i32x8;
typedef __attribute__((ext_vector_type(4)))  int       i32x4;
typedef _Float16 h16;

#if defined(__has_builtin)
#if __has_builtin(__builtin_amdgcn_tensor_load_to_lds) && \
    __has_builtin(__builtin_amdgcn_s_wait_tensorcnt)
#define USE_TDM 1
#endif
#endif

// ---------------------------------------------------------------------------
// Implicit-GEMM convolution with WMMA f16 (f32 accumulate).
//   Out[m, n] = sum_k Patch[m, k] * WmatT[n, k] + bias[n]
//   m -> (b, oh, ow)   k -> (r, s, cin) with cin fastest (NHWC-friendly)
//   Supports stride, zero-padding, and lhs-dilation 2 (ConvTranspose k6 s2 p2).
// Block: 256 threads = 8 waves; block tile 128(M) x 128(N); K-step 32.
// Wave tile 32x64 = 2x4 of 16x16 WMMA accumulators -> 8 v_wmma per K-step.
// B tile fetched by the Tensor Data Mover (TDM) when available: one
// tensor_load_to_lds per K-step issued by wave 0, LDS row padding done by the
// TDM pad engine (16 DW interval + 4 DW pad == 40-half row stride), fenced by
// s_wait_tensorcnt + the workgroup barrier.
// ---------------------------------------------------------------------------
struct ConvArgs {
  const h16*  act;     // NHWC f16 [B,H,W,Cin]
  const h16*  wmt;     // transposed weights [Npad][Kpad] f16, zero padded
  const float* bias;   // [Cout] or nullptr
  void*       out;     // NHWC, f16 or f32, leading dim = Cout
  int out_f32;
  int B, H, W, Cin, Cout, R, S, stride, pad, ldil;
  int OH, OW, Kpad, Npad, M, cin_log2;
};

#define A_STR 40
#define BT_STR 40

__global__ __launch_bounds__(256) void conv_wmma_k(ConvArgs a) {
  __shared__ __align__(16) h16 As[128 * A_STR];   // [row][k]   128x32 used
  __shared__ __align__(16) h16 Bt[128 * BT_STR];  // [n][k]     128x32 used

  const int tid  = threadIdx.x;
  const int lane = tid & 31;
  const int wave = tid >> 5;
  const int wm   = wave & 3;           // 32-row group
  const int wn   = wave >> 2;          // 64-col group
  const int n0   = blockIdx.x * 128;
  const int m0   = blockIdx.y * 128;
  const int OHOW = a.OH * a.OW;
  const int Kreal = a.R * a.S * a.Cin;
  const int Hv = (a.H - 1) * a.ldil + 1;
  const int Wv = (a.W - 1) * a.ldil + 1;
  const int cmask = a.Cin - 1;

  // ---- K-invariant per-thread staging descriptors ----
  // A: 512 chunks of 8 halves (128 rows x 4), 2 per thread.
  int   arow[2], akkb[2], aoh[2], aow[2];
  size_t abase[2];
  bool  avalid[2];
#pragma unroll
  for (int i = 0; i < 2; ++i) {
    int c   = i * 256 + tid;
    int row = c >> 2;
    arow[i] = row;
    akkb[i] = (c & 3) * 8;
    int m   = m0 + row;
    bool v  = m < a.M;
    int mm  = v ? m : 0;
    int b   = mm / OHOW;
    int rem = mm - b * OHOW;
    int oh  = rem / a.OW;
    int ow  = rem - oh * a.OW;
    avalid[i] = v;
    aoh[i]    = oh * a.stride - a.pad;
    aow[i]    = ow * a.stride - a.pad;
    abase[i]  = (size_t)b * a.H * a.W;           // pixel base
  }
#ifndef USE_TDM
  int brow[2], bkkb[2];
#pragma unroll
  for (int i = 0; i < 2; ++i) {
    int c = i * 256 + tid;
    brow[i] = c >> 2;
    bkkb[i] = (c & 3) * 8;
  }
#endif

  v8f acc[2][4];
  {
    v8f z = {};
#pragma unroll
    for (int g = 0; g < 2; ++g)
#pragma unroll
      for (int j = 0; j < 4; ++j) acc[g][j] = z;
  }

  const int lg  = (lane >= 16) ? 8 : 0;
  const int l15 = lane & 15;

  for (int k0 = 0; k0 < a.Kpad; k0 += 32) {
#ifdef USE_TDM
    // ---- stage B tile via Tensor Data Mover (one descriptor per K-step) ----
    if (wave == 0) {
      unsigned long long ga =
          (unsigned long long)(const void*)(a.wmt + (size_t)n0 * a.Kpad + k0);
      unsigned int lds = (unsigned int)(uintptr_t)(void*)Bt;
      u32x4 g0 = {};
      g0[0] = 1u;                                          // count=1 (valid D#)
      g0[1] = lds;                                         // lds_addr
      g0[2] = (unsigned int)ga;                            // global_addr[31:0]
      g0[3] = ((unsigned int)(ga >> 32) & 0x01FFFFFFu)     // global_addr[56:32]
              | (2u << 30);                                // type=2 ("image")
      i32x8 g1 = {};
      // data_size=1 (2B), pad_enable, pad_interval=3 (16 DW), pad_amount=3 (4 DW)
      g1[0] = (1 << 16) | (1 << 20) | (3 << 22) | (3 << 25);
      unsigned int td0 = (unsigned int)a.Kpad;             // tensor_dim0
      unsigned int td1 = 128u;                             // tensor_dim1
      g1[1] = (int)((td0 & 0xFFFFu) << 16);
      g1[2] = (int)((td0 >> 16) | ((td1 & 0xFFFFu) << 16));
      g1[3] = (int)((td1 >> 16) | (32u << 16));            // tile_dim0 = 32
      g1[4] = (int)128;                                    // tile_dim1 = 128
      g1[5] = (int)(unsigned int)a.Kpad;                   // dim0 stride
      i32x4 z4 = {};
#if __clang_major__ >= 23
      i32x8 z8 = {};
      __builtin_amdgcn_tensor_load_to_lds(g0, g1, z4, z4, z8, 0);
#else
      __builtin_amdgcn_tensor_load_to_lds(g0, g1, z4, z4, 0);
#endif
    }
#endif
    // ---- stage A tile: one b128 (or zero splat) per chunk ----
#pragma unroll
    for (int i = 0; i < 2; ++i) {
      int k = k0 + akkb[i];
      v8h vec = {};
      if (avalid[i] && k < Kreal) {
        int rs   = k >> a.cin_log2;
        int cin0 = k & cmask;
        int rr   = rs / a.S;
        int ss   = rs - rr * a.S;
        int iv   = aoh[i] + rr;
        int jv   = aow[i] + ss;
        bool ok = (iv >= 0) & (iv < Hv) & (jv >= 0) & (jv < Wv);
        if (a.ldil > 1) ok = ok && (((iv | jv) & 1) == 0);
        if (ok) {
          int ih = (a.ldil > 1) ? (iv >> 1) : iv;
          int iw = (a.ldil > 1) ? (jv >> 1) : jv;
          vec = *(const v8h*)(a.act +
                (((abase[i] + (size_t)ih * a.W + iw) << a.cin_log2) + cin0));
        }
      }
      *(v8h*)(As + arow[i] * A_STR + akkb[i]) = vec;
    }
#ifdef USE_TDM
    if (wave == 0) __builtin_amdgcn_s_wait_tensorcnt(0);
#else
    // ---- stage B tile: vector copies (weights pre-transposed & padded) ----
#pragma unroll
    for (int i = 0; i < 2; ++i) {
      const h16* src = a.wmt + (size_t)(n0 + brow[i]) * a.Kpad + k0 + bkkb[i];
      if (k0 + 32 < a.Kpad) __builtin_prefetch(src + 32, 0, 0);
      *(v8h*)(Bt + brow[i] * BT_STR + bkkb[i]) = *(const v8h*)src;
    }
#endif
    __syncthreads();

    // ---- fragments: two ds_read_b128 each (ISA 7.12.2 wave32 layouts) ----
    v16h af[2], bf[4];
#pragma unroll
    for (int g = 0; g < 2; ++g) {
      const h16* p = As + (wm * 32 + g * 16 + l15) * A_STR + lg;
      v8h lo = *(const v8h*)p;
      v8h hi = *(const v8h*)(p + 16);
      af[g] = __builtin_shufflevector(lo, hi, 0, 1, 2, 3, 4, 5, 6, 7,
                                      8, 9, 10, 11, 12, 13, 14, 15);
    }
#pragma unroll
    for (int j = 0; j < 4; ++j) {
      const h16* p = Bt + (wn * 64 + j * 16 + l15) * BT_STR + lg;
      v8h lo = *(const v8h*)p;
      v8h hi = *(const v8h*)(p + 16);
      bf[j] = __builtin_shufflevector(lo, hi, 0, 1, 2, 3, 4, 5, 6, 7,
                                      8, 9, 10, 11, 12, 13, 14, 15);
    }
#pragma unroll
    for (int j = 0; j < 4; ++j) {
      acc[0][j] = __builtin_amdgcn_wmma_f32_16x16x32_f16(false, af[0], false, bf[j], (short)0, acc[0][j], false, false);
      acc[1][j] = __builtin_amdgcn_wmma_f32_16x16x32_f16(false, af[1], false, bf[j], (short)0, acc[1][j], false, false);
    }
    __syncthreads();
  }

  // ---- epilogue: D layout (VGPR r: lanes0-15 M=r, lanes16-31 M=r+8) ----
  const int Nc    = l15;
  const int Mbase = (lane >= 16) ? 8 : 0;
#pragma unroll
  for (int mg = 0; mg < 2; ++mg)
#pragma unroll
    for (int ng = 0; ng < 4; ++ng)
#pragma unroll
      for (int r2 = 0; r2 < 8; ++r2) {
        int row = m0 + wm * 32 + mg * 16 + Mbase + r2;
        int col = n0 + wn * 64 + ng * 16 + Nc;
        if (row < a.M && col < a.Cout) {
          float v = acc[mg][ng][r2] + (a.bias ? a.bias[col] : 0.f);
          if (a.out_f32) ((float*)a.out)[(size_t)row * a.Cout + col] = v;
          else           ((h16*)a.out)[(size_t)row * a.Cout + col] = (h16)v;
        }
      }
}

// ---------------------------------------------------------------------------
// Weight re-layout: torch f32 -> transposed padded [Npad][Kpad] f16
// im2col K-order: k = (r*S + s)*Cin + cin
// ---------------------------------------------------------------------------
__global__ void prep_conv_w_k(const float* w, h16* out, int O, int Cin, int R,
                              int S, int Kpad, int Npad) {
  int t = blockIdx.x * 256 + threadIdx.x;
  if (t >= Kpad * Npad) return;
  int n = t / Kpad, k = t - (t / Kpad) * Kpad;
  float v = 0.f;
  if (k < R * S * Cin && n < O) {
    int cin = k % Cin, rs = k / Cin, r = rs / S, s = rs - (rs / S) * S;
    v = w[(((size_t)n * Cin + cin) * R + r) * S + s];   // OIHW
  }
  out[t] = (h16)v;
}

__global__ void prep_deconv_w_k(const float* w, h16* out, int O, int Cin, int R,
                                int S, int Kpad, int Npad) {
  int t = blockIdx.x * 256 + threadIdx.x;
  if (t >= Kpad * Npad) return;
  int n = t / Kpad, k = t - (t / Kpad) * Kpad;
  float v = 0.f;
  if (k < R * S * Cin && n < O) {
    int cin = k % Cin, rs = k / Cin, r = rs / S, s = rs - (rs / S) * S;
    // torch (in,out,kh,kw), spatially flipped, in/out transposed
    v = w[(((size_t)cin * O + n) * R + (R - 1 - r)) * S + (S - 1 - s)];
  }
  out[t] = (h16)v;
}

__global__ void prep_cb_k(const float* cb, h16* out) {
  // codebook [512,128] f32 -> [Npad=512][Kpad=128] f16 (same layout: direct cast)
  int t = blockIdx.x * 256 + threadIdx.x;
  if (t >= 512 * 128) return;
  out[t] = (h16)cb[t];
}

__global__ void csq_k(const float* cb, float* csq) {
  int n = blockIdx.x * 256 + threadIdx.x;
  if (n >= 512) return;
  float s = 0.f;
  for (int k = 0; k < 128; ++k) { float v = cb[(size_t)n * 128 + k]; s += v * v; }
  csq[n] = s;
}

// ---------------------------------------------------------------------------
// Elementwise / reduction kernels
// ---------------------------------------------------------------------------
__global__ void onehot_k(const int* x, h16* out, int n) {  // n pixels, C=8
  int t = blockIdx.x * 256 + threadIdx.x;
  if (t >= n) return;
  int cls = x[t];
#pragma unroll
  for (int c = 0; c < 8; ++c) out[(size_t)t * 8 + c] = (h16)(c == cls ? 1.f : 0.f);
}

__global__ void bn_stats_k(const h16* x, float* mean, float* var, int C, int Nsp) {
  __shared__ float s1[256], s2[256];
  int c = blockIdx.x, tid = threadIdx.x;
  float a = 0.f, b = 0.f;
  for (int i = tid; i < Nsp; i += 256) {
    float v = (float)x[(size_t)i * C + c];
    a += v; b += v * v;
  }
  s1[tid] = a; s2[tid] = b;
  __syncthreads();
  for (int off = 128; off > 0; off >>= 1) {
    if (tid < off) { s1[tid] += s1[tid + off]; s2[tid] += s2[tid + off]; }
    __syncthreads();
  }
  if (tid == 0) {
    float m = s1[0] / (float)Nsp;
    mean[c] = m;
    var[c]  = s2[0] / (float)Nsp - m * m;   // biased variance
  }
}

__global__ void bn_apply_k(const h16* x, h16* y, const float* mean, const float* var,
                           const float* g, const float* b, int C, long long n) {
  long long t = (long long)blockIdx.x * 256 + threadIdx.x;
  if (t >= n) return;
  int c = (int)(t % C);
  float v = ((float)x[t] - mean[c]) * rsqrtf(var[c] + 1e-5f) * g[c] + b[c];
  y[t] = (h16)v;
}

__global__ void elu_k(h16* x, long long n) {
  long long t = (long long)blockIdx.x * 256 + threadIdx.x;
  if (t >= n) return;
  float v = (float)x[t];
  x[t] = (h16)(v > 0.f ? v : (expf(v) - 1.f));
}

__global__ void glu_res_k(const h16* t3, h16* x, int C, long long Nsp) {
  long long t = (long long)blockIdx.x * 256 + threadIdx.x;
  if (t >= Nsp * C) return;
  long long i = t / C;
  int c = (int)(t % C);
  float aa = (float)t3[i * 2 * C + c];
  float gg = (float)t3[i * 2 * C + C + c];
  x[t] = (h16)(aa * (1.f / (1.f + expf(-gg))) + (float)x[t]);
}

__global__ void argmin_k(const float* scores, const float* csq, int* idx) {
  __shared__ float bm[256];
  __shared__ int   bi[256];
  int m = blockIdx.x, tid = threadIdx.x;
  float best = 3.4e38f; int bidx = 0;
  for (int n = tid; n < 512; n += 256) {
    float d = csq[n] - 2.f * scores[(size_t)m * 512 + n];
    if (d < best) { best = d; bidx = n; }
  }
  bm[tid] = best; bi[tid] = bidx;
  __syncthreads();
  for (int off = 128; off > 0; off >>= 1) {
    if (tid < off && bm[tid + off] < bm[tid]) { bm[tid] = bm[tid + off]; bi[tid] = bi[tid + off]; }
    __syncthreads();
  }
  if (tid == 0) idx[m] = bi[0];
}

__global__ void gather_diff_k(const int* idx, const float* cb, const h16* z,
                              h16* e, float* diff, float scale) {
  __shared__ float s[128];
  int m = blockIdx.x, c = threadIdx.x;  // 128 threads
  float ev = cb[(size_t)idx[m] * 128 + c];
  e[(size_t)m * 128 + c] = (h16)ev;
  float d = (float)z[(size_t)m * 128 + c] - ev;
  s[c] = d * d;
  __syncthreads();
  for (int off = 64; off > 0; off >>= 1) {
    if (c < off) s[c] += s[c + off];
    __syncthreads();
  }
  if (c == 0) atomicAdd(diff, s[0] * scale);
}

__global__ void zero_k(float* p) { if (threadIdx.x == 0 && blockIdx.x == 0) *p = 0.f; }

__global__ void to_nchw_k(const float* src, float* dst, int B, int C, int H, int W) {
  int t = blockIdx.x * 256 + threadIdx.x;
  int n = B * C * H * W;
  if (t >= n) return;
  int w = t % W, h = (t / W) % H, c = (t / (W * H)) % C, b = t / (W * H * C);
  dst[t] = src[(((size_t)b * H + h) * W + w) * C + c];
}

// ---------------------------------------------------------------------------
// Host orchestration
// ---------------------------------------------------------------------------
extern "C" void kernel_launch(void* const* d_in, const int* in_sizes, int n_in,
                              void* d_out, int out_size, void* d_ws, size_t ws_size,
                              hipStream_t stream) {
  (void)in_sizes; (void)n_in; (void)out_size;
  const int* x_int = (const int*)d_in[0];
  auto F = [&](int i) { return (const float*)d_in[i]; };

  // ---- workspace bump allocator ----
  char* ws = (char*)d_ws;
  size_t off = 0;
  auto alloc = [&](size_t bytes) -> char* {
    char* p = ws + off;
    off += (bytes + 255) & ~(size_t)255;
    return p;
  };
  const size_t ACT_C  = (size_t)4 * 96 * 96 * 128 * sizeof(h16);
  h16*   P0     = (h16*)alloc(ACT_C);
  h16*   P1     = (h16*)alloc(ACT_C);
  h16*   P2     = (h16*)alloc(ACT_C);
  h16*   PD     = (h16*)alloc(ACT_C * 2);             // 2C buffer
  h16*   W0     = (h16*)alloc(2u * 1024 * 1024);      // weight slot
  h16*   WCB    = (h16*)alloc(128 * 512 * sizeof(h16));
  float* STATS  = (float*)alloc(512 * sizeof(float)); // mean[256] | var[256]
  float* SCORES = (float*)alloc((size_t)576 * 512 * sizeof(float));
  float* CSQ    = (float*)alloc(512 * sizeof(float));
  int*   IDX    = (int*)alloc(576 * sizeof(int));
  float* OUTF   = (float*)alloc((size_t)4 * 96 * 96 * 8 * sizeof(float));
  if (off > ws_size) return;

  float* dout = (float*)d_out;
  float* diff_slot = dout + (size_t)4 * 8 * 96 * 96;

  auto prep_w = [&](const float* w, int O, int Cin, int R, int S, h16* dst, bool deconv) {
    int Kpad = ((R * S * Cin) + 31) & ~31;
    int Npad = (O + 127) & ~127;
    int n = Kpad * Npad;
    if (deconv) prep_deconv_w_k<<<(n + 255) / 256, 256, 0, stream>>>(w, dst, O, Cin, R, S, Kpad, Npad);
    else        prep_conv_w_k<<<(n + 255) / 256, 256, 0, stream>>>(w, dst, O, Cin, R, S, Kpad, Npad);
  };

  auto conv = [&](const h16* act, int B, int H, int W, int Cin, int Cout, int R, int S,
                  int st, int pad, int ldil, const h16* wmt, const float* bias,
                  void* out, int out_f32, int& OH, int& OW) {
    OH = ((H - 1) * ldil + 1 + 2 * pad - R) / st + 1;
    OW = ((W - 1) * ldil + 1 + 2 * pad - S) / st + 1;
    ConvArgs a;
    a.act = act; a.wmt = wmt; a.bias = bias; a.out = out; a.out_f32 = out_f32;
    a.B = B; a.H = H; a.W = W; a.Cin = Cin; a.Cout = Cout;
    a.R = R; a.S = S; a.stride = st; a.pad = pad; a.ldil = ldil;
    a.OH = OH; a.OW = OW;
    a.Kpad = ((R * S * Cin) + 31) & ~31;
    a.Npad = (Cout + 127) & ~127;
    a.M = B * OH * OW;
    a.cin_log2 = __builtin_ctz((unsigned)Cin);
    dim3 grid(a.Npad / 128, (a.M + 127) / 128);
    conv_wmma_k<<<grid, 256, 0, stream>>>(a);
  };

  auto bn = [&](const h16* src, h16* dst, int C, int Nsp, const float* g, const float* b) {
    bn_stats_k<<<C, 256, 0, stream>>>(src, STATS, STATS + 256, C, Nsp);
    long long n = (long long)Nsp * C;
    bn_apply_k<<<(unsigned)((n + 255) / 256), 256, 0, stream>>>(src, dst, STATS, STATS + 256, g, b, C, n);
  };
  auto elu = [&](h16* buf, long long n) {
    elu_k<<<(unsigned)((n + 255) / 256), 256, 0, stream>>>(buf, n);
  };

  // gated resnet: cur updated in place; P = {b1,b2,bn1_b,bn1_g,bn2_b,bn2_g,w1,w2}
  auto resnet = [&](const float* const* P, h16* cur, h16* t1, h16* t2, h16* t3,
                    int B, int H, int W) {
    int Nsp = B * H * W, oh, ow;
    bn(cur, t1, 128, Nsp, P[3], P[2]);
    elu(t1, (long long)Nsp * 128);
    prep_w(P[6], 128, 128, 3, 3, W0, false);
    conv(t1, B, H, W, 128, 128, 3, 3, 1, 1, 1, W0, P[0], t2, 0, oh, ow);
    elu(t2, (long long)Nsp * 128);
    prep_w(P[7], 256, 128, 3, 3, W0, false);
    conv(t2, B, H, W, 128, 256, 3, 3, 1, 1, 1, W0, P[1], t3, 0, oh, ow);
    bn(t3, t3, 256, Nsp, P[5], P[4]);
    long long n = (long long)Nsp * 128;
    glu_res_k<<<(unsigned)((n + 255) / 256), 256, 0, stream>>>(t3, cur, 128, Nsp);
  };

  // ---- input tensor indices (jax pytree: dict keys sorted, lists in order) ----
  // 0:x 1:codebook | dec_blocks 2..109 | dec_final_res 110 | dec_in_b 118
  // dec_in_bn_b 119 dec_in_bn_g 120 dec_in_w 121 dec_out_b 122 dec_out_w 123
  // enc_blocks 124..231 | enc_final_res 232 | enc_in_b 240 enc_in_bn_b 241
  // enc_in_bn_g 242 enc_in_w 243

  h16 *cur = P0, *tA = P1, *tB = P2;
  int H = 96, W = 96, oh, ow;

  // ===== Encoder =====
  onehot_k<<<(4 * 96 * 96 + 255) / 256, 256, 0, stream>>>(x_int, tA, 4 * 96 * 96);
  prep_w(F(243), 128, 8, 7, 7, W0, false);
  conv(tA, 4, 96, 96, 8, 128, 7, 7, 1, 0, 1, W0, F(240), cur, 0, oh, ow);   // -> 90x90
  H = oh; W = ow;
  bn(cur, cur, 128, 4 * H * W, F(242), F(241));

  for (int blk = 0; blk < 3; ++blk) {
    int base = 124 + blk * 36;  // bn_b, bn_g, down_b, down_w, res[4]x8
    for (int r = 0; r < 4; ++r) {
      const float* P[8];
      for (int j = 0; j < 8; ++j) P[j] = F(base + 4 + r * 8 + j);
      resnet(P, cur, tA, tB, PD, 4, H, W);
    }
    prep_w(F(base + 3), 128, 128, 5, 5, W0, false);
    conv(cur, 4, H, W, 128, 128, 5, 5, 2, 2, 1, W0, F(base + 2), tA, 0, oh, ow);
    bn(tA, tA, 128, 4 * oh * ow, F(base + 1), F(base + 0));
    h16* t = cur; cur = tA; tA = t;
    H = oh; W = ow;                              // 90 -> 45 -> 23 -> 12
  }
  {
    const float* P[8];
    for (int j = 0; j < 8; ++j) P[j] = F(232 + j);
    resnet(P, cur, tA, tB, PD, 4, H, W);         // z = cur (12x12)
  }

  // ===== Quantize: dist = |z|^2 - 2 z.c + |c|^2 ; argmin over 512 =====
  prep_cb_k<<<(512 * 128 + 255) / 256, 256, 0, stream>>>(F(1), WCB);
  csq_k<<<2, 256, 0, stream>>>(F(1), CSQ);
  conv(cur, 4, H, W, 128, 512, 1, 1, 1, 0, 1, WCB, nullptr, SCORES, 1, oh, ow);
  argmin_k<<<576, 256, 0, stream>>>(SCORES, CSQ, IDX);
  zero_k<<<1, 1, 0, stream>>>(diff_slot);
  gather_diff_k<<<576, 128, 0, stream>>>(IDX, F(1), cur, tA, diff_slot,
                                         2.f / (576.f * 128.f));
  { h16* t = cur; cur = tA; tA = t; }            // cur = e_st

  // ===== Decoder =====
  prep_w(F(121), 128, 128, 3, 3, W0, false);
  conv(cur, 4, H, W, 128, 128, 3, 3, 1, 1, 1, W0, F(118), tA, 0, oh, ow);
  bn(tA, tA, 128, 4 * H * W, F(120), F(119));
  { h16* t = cur; cur = tA; tA = t; }

  for (int blk = 0; blk < 3; ++blk) {
    int base = 2 + blk * 36;  // bn_b, bn_g, res[4]x8, up_b, up_w
    for (int r = 0; r < 4; ++r) {
      const float* P[8];
      for (int j = 0; j < 8; ++j) P[j] = F(base + 2 + r * 8 + j);
      resnet(P, cur, tA, tB, PD, 4, H, W);
    }
    prep_w(F(base + 35), 128, 128, 6, 6, W0, true);   // ConvTranspose k6 s2 p2
    conv(cur, 4, H, W, 128, 128, 6, 6, 1, 3, 2, W0, F(base + 34), tA, 0, oh, ow);
    bn(tA, tA, 128, 4 * oh * ow, F(base + 1), F(base + 0));
    h16* t = cur; cur = tA; tA = t;
    H = oh; W = ow;                              // 12 -> 24 -> 48 -> 96
  }
  {
    const float* P[8];
    for (int j = 0; j < 8; ++j) P[j] = F(110 + j);
    resnet(P, cur, tA, tB, PD, 4, H, W);
  }

  // ===== Output head: x_tilde = conv1x1(elu(out)) =====
  elu(cur, (long long)4 * H * W * 128);
  prep_w(F(123), 8, 128, 1, 1, W0, false);
  conv(cur, 4, H, W, 128, 8, 1, 1, 1, 0, 1, W0, F(122), OUTF, 1, oh, ow);
  to_nchw_k<<<(4 * 8 * 96 * 96 + 255) / 256, 256, 0, stream>>>(OUTF, dout, 4, 8, 96, 96);
}
